// CausalSelfAttention_10033043604042
// MI455X (gfx1250) — matrix operand order repaired and checked by reading, not compiled, not run on hardware
//
#include <hip/hip_runtime.h>
#include <hip/hip_bf16.h>

#define BATCH 4
#define SEQ   2048
#define EMB   1024
#define NH    16
#define HD    64
#define N3E   3072
#define MTOT  (BATCH*SEQ)   // 8192

typedef __attribute__((ext_vector_type(16))) __bf16 v16bf;
typedef __attribute__((ext_vector_type(8)))  float  v8f;
typedef __attribute__((ext_vector_type(4)))  int    v4i;
typedef unsigned short ushortT;

#ifndef __has_builtin
#define __has_builtin(x) 0
#endif
#if __has_builtin(__builtin_amdgcn_global_load_async_to_lds_b128)
#define ASYNC_COPY 1
#else
#define ASYNC_COPY 0
#endif

#if ASYNC_COPY
typedef __attribute__((address_space(3))) v4i lds_v4i;
__device__ __forceinline__ void async_b128(const void* g, void* l) {
  __builtin_amdgcn_global_load_async_to_lds_b128((v4i*)g, (lds_v4i*)l, 0, 0);
}
__device__ __forceinline__ void wait_async0() {
#if __has_builtin(__builtin_amdgcn_s_wait_asynccnt)
  __builtin_amdgcn_s_wait_asynccnt(0);
#else
  asm volatile("s_wait_asynccnt 0" ::: "memory");
#endif
}
#endif

__device__ __forceinline__ ushortT bf_hi(float x) {
  __bf16 h = (__bf16)x;
  return __builtin_bit_cast(ushortT, h);
}
__device__ __forceinline__ ushortT bf_lo(float x) {
  __bf16 h = (__bf16)x;
  __bf16 l = (__bf16)(x - (float)h);
  return __builtin_bit_cast(ushortT, l);
}

// assemble a 16-element bf16 fragment from two 16B chunks (LDS or global)
__device__ __forceinline__ v16bf ld_frag(const ushortT* p0, const ushortT* p1) {
  union { v16bf v; uint4 q[2]; } u;
  u.q[0] = *(const uint4*)p0;
  u.q[1] = *(const uint4*)p1;
  return u.v;
}

// D = A*B + C with bf16 error compensation: hi*hi + hi*lo + lo*hi
__device__ __forceinline__ v8f wmma_bf16x3(v16bf a_hi, v16bf a_lo,
                                           v16bf b_hi, v16bf b_lo, v8f acc) {
  acc = __builtin_amdgcn_wmma_f32_16x16x32_bf16(false, a_hi, false, b_hi, (short)0, acc, false, false);
  acc = __builtin_amdgcn_wmma_f32_16x16x32_bf16(false, a_hi, false, b_lo, (short)0, acc, false, false);
  acc = __builtin_amdgcn_wmma_f32_16x16x32_bf16(false, a_lo, false, b_hi, (short)0, acc, false, false);
  return acc;
}

// ---------------------------------------------------------------------------
// pack_split: elementwise f32 -> (hi,lo) bf16, same layout. 4 elems/thread.
// ---------------------------------------------------------------------------
__global__ __launch_bounds__(256) void pack_split(const float* __restrict__ src,
    ushortT* __restrict__ hi, ushortT* __restrict__ lo)
{
  size_t i = ((size_t)blockIdx.x * 256 + threadIdx.x) * 4;
  float4 v = *(const float4*)(src + i);
  ushort4 h, l;
  h.x = bf_hi(v.x); l.x = bf_lo(v.x);
  h.y = bf_hi(v.y); l.y = bf_lo(v.y);
  h.z = bf_hi(v.z); l.z = bf_lo(v.z);
  h.w = bf_hi(v.w); l.w = bf_lo(v.w);
  *(ushort4*)(hi + i) = h;
  *(ushort4*)(lo + i) = l;
}

// ---------------------------------------------------------------------------
// pack_wT: W[1024][N] f32 -> WT[N][1024] bf16 hi/lo (32x32 LDS tile transpose)
// grid: (1024/32, N/32), block 256
// ---------------------------------------------------------------------------
__global__ __launch_bounds__(256) void pack_wT(const float* __restrict__ W,
    ushortT* __restrict__ WTh, ushortT* __restrict__ WTl, int N)
{
  __shared__ float tile[32][33];
  const int tid = threadIdx.x;
  const int k0 = blockIdx.x * 32, n0 = blockIdx.y * 32;
  {
    int r = tid >> 3, c4 = (tid & 7) * 4;       // 32 rows x 8 float4
    float4 v = *(const float4*)(W + (size_t)(k0 + r) * N + n0 + c4);
    tile[r][c4+0]=v.x; tile[r][c4+1]=v.y; tile[r][c4+2]=v.z; tile[r][c4+3]=v.w;
  }
  __syncthreads();
  {
    int rw = tid >> 3, cw4 = (tid & 7) * 4;     // out row = n, 4 consecutive k
    ushort4 h, l;
    float a = tile[cw4+0][rw], b = tile[cw4+1][rw], c = tile[cw4+2][rw], d = tile[cw4+3][rw];
    h.x = bf_hi(a); l.x = bf_lo(a);
    h.y = bf_hi(b); l.y = bf_lo(b);
    h.z = bf_hi(c); l.z = bf_lo(c);
    h.w = bf_hi(d); l.w = bf_lo(d);
    size_t o = (size_t)(n0 + rw) * EMB + k0 + cw4;
    *(ushort4*)(WTh + o) = h;
    *(ushort4*)(WTl + o) = l;
  }
}

// ---------------------------------------------------------------------------
// trans_v: V (bh,s,d) f32 -> Vt (bh,d,s) bf16 hi/lo. 32x32 LDS tile.
// grid: (B*H, SEQ/32, HD/32), block 256
// ---------------------------------------------------------------------------
__global__ __launch_bounds__(256) void trans_v(const float* __restrict__ V,
    ushortT* __restrict__ Vth, ushortT* __restrict__ Vtl)
{
  __shared__ float tile[32][33];
  const int tid = threadIdx.x;
  const int bh = blockIdx.x;
  const int s0 = blockIdx.y * 32, d0 = blockIdx.z * 32;
  const float* Vb = V + (size_t)bh * SEQ * HD;
  {
    int r = tid >> 3, c4 = (tid & 7) * 4;
    float4 v = *(const float4*)(Vb + (size_t)(s0 + r) * HD + d0 + c4);
    tile[r][c4+0]=v.x; tile[r][c4+1]=v.y; tile[r][c4+2]=v.z; tile[r][c4+3]=v.w;
  }
  __syncthreads();
  {
    int rw = tid >> 3, cw4 = (tid & 7) * 4;     // out row = d, 4 consecutive s
    ushort4 h, l;
    float a = tile[cw4+0][rw], b = tile[cw4+1][rw], c = tile[cw4+2][rw], d = tile[cw4+3][rw];
    h.x = bf_hi(a); l.x = bf_lo(a);
    h.y = bf_hi(b); l.y = bf_lo(b);
    h.z = bf_hi(c); l.z = bf_lo(c);
    h.w = bf_hi(d); l.w = bf_lo(d);
    size_t o = (size_t)bh * HD * SEQ + (size_t)(d0 + rw) * SEQ + s0 + cw4;
    *(ushort4*)(Vth + o) = h;
    *(ushort4*)(Vtl + o) = l;
  }
}

// ---------------------------------------------------------------------------
// GEMM: C[M,N] = A[M,1024] * W[1024,N] + bias[N], pre-split bf16 operands.
// A: (M,1024) bf16 hi/lo row-major. B: W^T (N,1024) bf16 hi/lo row-major.
// mode 0: scatter [3][H][D] cols -> Q f32, Khi/Klo bf16, V f32 (all (B,H,S,D))
// mode 1: dense f32 Fo[M,N]
// Block 256 (8 waves), tile 128x128, K-step 32, 24 WMMA/wave/k-step.
// Async global->LDS staging (double-buffered) when available.
// ---------------------------------------------------------------------------
__global__ __launch_bounds__(256) void gemm_bf16(
    const ushortT* __restrict__ Ahi, const ushortT* __restrict__ Alo,
    const ushortT* __restrict__ Bhi, const ushortT* __restrict__ Blo,
    const float* __restrict__ bias,
    float* __restrict__ Qo, ushortT* __restrict__ Kh, ushortT* __restrict__ Kl,
    float* __restrict__ Vo, float* __restrict__ Fo, int N, int mode)
{
  // padded rows: 40 ushorts = 80B (16B-aligned, conflict-free b128 lanes)
  __shared__ __align__(16) ushortT Ah[2][128][40], Al[2][128][40];
  __shared__ __align__(16) ushortT Bh[2][128][40], Bl[2][128][40];

  const int tid  = threadIdx.x;
  const int wave = tid >> 5, lane = tid & 31;
  const int half = lane >> 4, lm = lane & 15;
  const int rm0  = blockIdx.x * 128;
  const int n0   = blockIdx.y * 128;

  // staging coordinates for this thread (2 chunks per operand tile)
  const int r0 = tid >> 2, c0 = (tid & 3) * 8;          // f = tid
  const int r1 = (tid + 256) >> 2, c1 = ((tid + 256) & 3) * 8;

  v8f acc[8];
#pragma unroll
  for (int c = 0; c < 8; ++c) acc[c] = (v8f){0,0,0,0,0,0,0,0};

#if ASYNC_COPY
  // prologue: async-stage k0=0 into buffer 0
  {
    size_t a0 = (size_t)(rm0 + r0) * EMB + c0, a1 = (size_t)(rm0 + r1) * EMB + c1;
    size_t b0 = (size_t)(n0 + r0) * EMB + c0, b1 = (size_t)(n0 + r1) * EMB + c1;
    async_b128(Ahi + a0, &Ah[0][r0][c0]); async_b128(Ahi + a1, &Ah[0][r1][c1]);
    async_b128(Alo + a0, &Al[0][r0][c0]); async_b128(Alo + a1, &Al[0][r1][c1]);
    async_b128(Bhi + b0, &Bh[0][r0][c0]); async_b128(Bhi + b1, &Bh[0][r1][c1]);
    async_b128(Blo + b0, &Bl[0][r0][c0]); async_b128(Blo + b1, &Bl[0][r1][c1]);
  }
  wait_async0();
  __syncthreads();
#endif

  int ibuf = 0;
  for (int k0 = 0; k0 < EMB; k0 += 32, ibuf ^= 1) {
#if ASYNC_COPY
    // issue async staging of the next k-step into the other buffer
    if (k0 + 32 < EMB) {
      int nb = ibuf ^ 1, kn = k0 + 32;
      size_t a0 = (size_t)(rm0 + r0) * EMB + kn + c0, a1 = (size_t)(rm0 + r1) * EMB + kn + c1;
      size_t b0 = (size_t)(n0 + r0) * EMB + kn + c0, b1 = (size_t)(n0 + r1) * EMB + kn + c1;
      async_b128(Ahi + a0, &Ah[nb][r0][c0]); async_b128(Ahi + a1, &Ah[nb][r1][c1]);
      async_b128(Alo + a0, &Al[nb][r0][c0]); async_b128(Alo + a1, &Al[nb][r1][c1]);
      async_b128(Bhi + b0, &Bh[nb][r0][c0]); async_b128(Bhi + b1, &Bh[nb][r1][c1]);
      async_b128(Blo + b0, &Bl[nb][r0][c0]); async_b128(Blo + b1, &Bl[nb][r1][c1]);
    }
#else
    ibuf = 0;
    __syncthreads();
    {
      size_t a0 = (size_t)(rm0 + r0) * EMB + k0 + c0, a1 = (size_t)(rm0 + r1) * EMB + k0 + c1;
      size_t b0 = (size_t)(n0 + r0) * EMB + k0 + c0, b1 = (size_t)(n0 + r1) * EMB + k0 + c1;
      uint4 t0 = *(const uint4*)(Ahi + a0), t1 = *(const uint4*)(Ahi + a1);
      uint4 t2 = *(const uint4*)(Alo + a0), t3 = *(const uint4*)(Alo + a1);
      uint4 t4 = *(const uint4*)(Bhi + b0), t5 = *(const uint4*)(Bhi + b1);
      uint4 t6 = *(const uint4*)(Blo + b0), t7 = *(const uint4*)(Blo + b1);
      *(uint4*)&Ah[0][r0][c0] = t0; *(uint4*)&Ah[0][r1][c1] = t1;
      *(uint4*)&Al[0][r0][c0] = t2; *(uint4*)&Al[0][r1][c1] = t3;
      *(uint4*)&Bh[0][r0][c0] = t4; *(uint4*)&Bh[0][r1][c1] = t5;
      *(uint4*)&Bl[0][r0][c0] = t6; *(uint4*)&Bl[0][r1][c1] = t7;
    }
    __syncthreads();
#endif

    // A fragment: contiguous K-run [16h..16h+15]
    const int ar = wave*16 + lm;
    v16bf a_hi = ld_frag(&Ah[ibuf][ar][16*half], &Ah[ibuf][ar][16*half + 8]);
    v16bf a_lo = ld_frag(&Al[ibuf][ar][16*half], &Al[ibuf][ar][16*half + 8]);
#pragma unroll
    for (int c = 0; c < 8; ++c) {
      // B fragment: contiguous K-run [16h..16h+15] of row n
      const int br = c*16 + lm;
      v16bf b_hi = ld_frag(&Bh[ibuf][br][16*half], &Bh[ibuf][br][16*half + 8]);
      v16bf b_lo = ld_frag(&Bl[ibuf][br][16*half], &Bl[ibuf][br][16*half + 8]);
      acc[c] = wmma_bf16x3(a_hi, a_lo, b_hi, b_lo, acc[c]);
    }

#if ASYNC_COPY
    wait_async0();      // this wave's async writes for next step complete
    __syncthreads();    // everyone done reading ibuf / writing ibuf^1
#endif
  }

  // epilogue: C/D layout M = r + 8*half, N = lm
#pragma unroll
  for (int c = 0; c < 8; ++c) {
#pragma unroll
    for (int r = 0; r < 8; ++r) {
      int m = rm0 + wave*16 + r + 8*half;
      int n = n0 + c*16 + lm;
      float v = acc[c][r] + bias[n];
      if (mode == 0) {
        int bb = m >> 11, s = m & (SEQ-1);
        int which = n >> 10, nn = n & 1023;
        int hh = nn >> 6, d = nn & 63;
        size_t idx = (((size_t)bb*NH + hh)*SEQ + s)*HD + d;
        if (which == 0)      Qo[idx] = v;
        else if (which == 1) { Kh[idx] = bf_hi(v); Kl[idx] = bf_lo(v); }
        else                 Vo[idx] = v;
      } else {
        Fo[(size_t)m*N + n] = v;
      }
    }
  }
}

// ---------------------------------------------------------------------------
// Flash attention with ALiBi + causal mask. Waves fully independent:
// block 256 (8 waves), each wave owns 16 q-rows, 32 keys per iteration.
// K fragments: direct global b128 from Khi/Klo (b,h,s,d).
// V fragments: direct global b128 from Vthi/Vtlo (b,h,d,s).
// Output written pre-split (AOhi/AOlo) for the final projection GEMM.
// ---------------------------------------------------------------------------
__global__ __launch_bounds__(256) void attn_kernel(
    const float*  __restrict__ Q,
    const ushortT* __restrict__ Khi, const ushortT* __restrict__ Klo,
    const ushortT* __restrict__ Vth, const ushortT* __restrict__ Vtl,
    ushortT* __restrict__ AOh, ushortT* __restrict__ AOl)
{
  __shared__ float Ps[8][16][34];   // per-wave P transpose buffer

  const int tid  = threadIdx.x;
  const int wave = tid >> 5, lane = tid & 31;
  const int half = lane >> 4, lm = lane & 15;
  const int bh = blockIdx.x;
  const int b  = bh >> 4, h = bh & 15;
  const int qrow0 = blockIdx.y * 128 + wave * 16;
  const float slope = exp2f(-0.5f * (float)(h + 1));   // (2^8)^(1/16) = 2^0.5

  const float*   Qb  = Q   + (size_t)bh * SEQ * HD;
  const ushortT* Khb = Khi + (size_t)bh * SEQ * HD;
  const ushortT* Klb = Klo + (size_t)bh * SEQ * HD;
  const ushortT* Vhb = Vth + (size_t)bh * HD * SEQ;
  const ushortT* Vlb = Vtl + (size_t)bh * HD * SEQ;

  // Q A-fragments for the two d-steps, pre-scaled by 1/sqrt(D)=0.125
  v16bf qh[2], ql[2];
#pragma unroll
  for (int t = 0; t < 2; ++t) {
#pragma unroll
    for (int e = 0; e < 16; ++e) {
      int d = 32*t + 16*half + e;
      float x = Qb[(size_t)(qrow0 + lm)*HD + d] * 0.125f;
      __bf16 hh = (__bf16)x;
      qh[t][e] = hh;
      ql[t][e] = (__bf16)(x - (float)hh);
    }
  }

  v8f o[4];
#pragma unroll
  for (int c = 0; c < 4; ++c) o[c] = (v8f){0,0,0,0,0,0,0,0};
  float mi[8], li[8];
#pragma unroll
  for (int r = 0; r < 8; ++r) { mi[r] = -3.0e38f; li[r] = 0.f; }

  const int njb = qrow0/32 + 1;   // causal key-block range for this wave
  for (int j = 0; j < njb; ++j) {
    const int kbase = j*32;

    // scores: two 16x16 tiles (keys kbase..+15, kbase+16..+31)
    v8f s[2];
    s[0] = (v8f){0,0,0,0,0,0,0,0};
    s[1] = (v8f){0,0,0,0,0,0,0,0};
#pragma unroll
    for (int t = 0; t < 2; ++t) {
#pragma unroll
      for (int u = 0; u < 2; ++u) {
        // B fragment: row key=kbase+16u+lm, contiguous d-run [32t+16h .. +15]
        size_t ko = (size_t)(kbase + 16*u + lm)*HD + 32*t + 16*half;
        v16bf kbh = ld_frag(Khb + ko, Khb + ko + 8);
        v16bf kbl = ld_frag(Klb + ko, Klb + ko + 8);
        s[u] = wmma_bf16x3(qh[t], ql[t], kbh, kbl, s[u]);
      }
    }

    // ALiBi + causal + online softmax (row = r + 8*half, col = lm per tile)
    float pnew[2][8];
#pragma unroll
    for (int r = 0; r < 8; ++r) {
      int row = qrow0 + r + 8*half;
      float rmax = -3.0e38f;
      float sc[2];
#pragma unroll
      for (int u = 0; u < 2; ++u) {
        int col = kbase + 16*u + lm;
        float x = s[u][r] + slope * (float)(col - row);
        if (col > row) x = -1.0e9f;
        sc[u] = x;
        rmax = fmaxf(rmax, x);
      }
#pragma unroll
      for (int off = 1; off < 16; off <<= 1)
        rmax = fmaxf(rmax, __shfl_xor(rmax, off, 32));
      float mnew  = fmaxf(mi[r], rmax);
      float scale = __expf(mi[r] - mnew);
      float rs = 0.f;
#pragma unroll
      for (int u = 0; u < 2; ++u) {
        float p = __expf(sc[u] - mnew);
        pnew[u][r] = p;
        rs += p;
      }
#pragma unroll
      for (int off = 1; off < 16; off <<= 1)
        rs += __shfl_xor(rs, off, 32);
      li[r] = li[r]*scale + rs;
      mi[r] = mnew;
#pragma unroll
      for (int c = 0; c < 4; ++c) o[c][r] *= scale;
    }

    // transpose P through per-wave LDS (C layout -> A layout)
#pragma unroll
    for (int r = 0; r < 8; ++r) {
#pragma unroll
      for (int u = 0; u < 2; ++u)
        Ps[wave][r + 8*half][16*u + lm] = pnew[u][r];
    }
    asm volatile("s_wait_dscnt 0" ::: "memory");   // order wave-local LDS RAW

    v16bf ph, pl;   // A fragment of P (16 rows x 32 keys)
#pragma unroll
    for (int e = 0; e < 16; ++e) {
      int kk = 16*half + e;
      float x = Ps[wave][lm][kk];
      __bf16 hh = (__bf16)x;
      ph[e] = hh;
      pl[e] = (__bf16)(x - (float)hh);
    }
#pragma unroll
    for (int c = 0; c < 4; ++c) {
      // B fragment: row d=c*16+lm of Vt, contiguous key-run [kbase+16h .. +15]
      size_t vo = (size_t)(c*16 + lm)*SEQ + kbase + 16*half;
      v16bf vh = ld_frag(Vhb + vo, Vhb + vo + 8);
      v16bf vl = ld_frag(Vlb + vo, Vlb + vo + 8);
      o[c] = wmma_bf16x3(ph, pl, vh, vl, o[c]);
    }
  }

  // normalize and write (B,S,E) pre-split bf16 for the output projection
#pragma unroll
  for (int c = 0; c < 4; ++c) {
#pragma unroll
    for (int r = 0; r < 8; ++r) {
      int row = qrow0 + r + 8*half;
      int d = c*16 + lm;
      float v = o[c][r] / li[r];
      size_t idx = ((size_t)b*SEQ + row)*EMB + h*HD + d;
      AOh[idx] = bf_hi(v);
      AOl[idx] = bf_lo(v);
    }
  }
}

// ---------------------------------------------------------------------------
extern "C" void kernel_launch(void* const* d_in, const int* in_sizes, int n_in,
                              void* d_out, int out_size, void* d_ws, size_t ws_size,
                              hipStream_t stream) {
  const float* x     = (const float*)d_in[0];
  // d_in[1] = mask (implicit causal; unused)
  const float* W_qkv = (const float*)d_in[2];
  const float* b_qkv = (const float*)d_in[3];
  const float* W_out = (const float*)d_in[4];
  const float* b_out = (const float*)d_in[5];

  char* w = (char*)d_ws;
  auto take = [&](size_t bytes) { char* p = w; w += (bytes + 255) & ~(size_t)255; return p; };
  const size_t NE  = (size_t)MTOT * EMB;            // 8,388,608 elems
  const size_t PH  = (size_t)BATCH * NH * SEQ * HD; // 8,388,608 elems

  ushortT* Xhi = (ushortT*)take(NE * 2);
  ushortT* Xlo = (ushortT*)take(NE * 2);
  ushortT* WTh = (ushortT*)take((size_t)N3E * EMB * 2);
  ushortT* WTl = (ushortT*)take((size_t)N3E * EMB * 2);
  ushortT* WoTh = (ushortT*)take((size_t)EMB * EMB * 2);
  ushortT* WoTl = (ushortT*)take((size_t)EMB * EMB * 2);
  float*   Qf  = (float*)take(PH * 4);
  ushortT* Khi = (ushortT*)take(PH * 2);
  ushortT* Klo = (ushortT*)take(PH * 2);
  float*   Vf  = (float*)take(PH * 4);
  ushortT* Vth = (ushortT*)take(PH * 2);
  ushortT* Vtl = (ushortT*)take(PH * 2);
  ushortT* AOh = (ushortT*)take(NE * 2);
  ushortT* AOl = (ushortT*)take(NE * 2);

  // 1) pre-split x; pre-split+transpose weights
  pack_split<<<dim3(NE / (256*4)), 256, 0, stream>>>(x, Xhi, Xlo);
  pack_wT<<<dim3(EMB/32, N3E/32), 256, 0, stream>>>(W_qkv, WTh, WTl, N3E);
  pack_wT<<<dim3(EMB/32, EMB/32), 256, 0, stream>>>(W_out, WoTh, WoTl, EMB);

  // 2) QKV projection (K written pre-split)
  gemm_bf16<<<dim3(MTOT/128, N3E/128), 256, 0, stream>>>(
      Xhi, Xlo, WTh, WTl, b_qkv, Qf, Khi, Klo, Vf, nullptr, N3E, 0);

  // 3) V transpose + split
  trans_v<<<dim3(BATCH*NH, SEQ/32, HD/32), 256, 0, stream>>>(Vf, Vth, Vtl);

  // 4) attention
  attn_kernel<<<dim3(BATCH*NH, SEQ/128), 256, 0, stream>>>(
      Qf, Khi, Klo, Vth, Vtl, AOh, AOl);

  // 5) output projection
  gemm_bf16<<<dim3(MTOT/128, EMB/128), 256, 0, stream>>>(
      AOh, AOl, WoTh, WoTl, b_out, nullptr, nullptr, nullptr, nullptr,
      (float*)d_out, EMB, 1);
}
